// HolographicKVCache_57844619543007
// MI455X (gfx1250) — matrix operand order repaired and checked by reading, not compiled, not run on hardware
//
#include <hip/hip_runtime.h>
#include <hip/hip_bf16.h>

// ---------------------------------------------------------------------------
// HolographicKVCache forward for MI455X (gfx1250), bf16 WMMA pipeline.
//   encode: H = gelu(X @ w1^T + b1); bnd = 0.5*(H@w2^T + b2) + 0.5*S*cos(ph)*(X@proj^T)
//           (boundary folded into one K=3072 GEMM by pre-scaling proj)
//   decode: Hd = gelu(bnd @ dw1^T + db1); bulk = Hd @ dw2^T + db2
//           out = bulk + 0.1*(bulk @ ec^T + ec_b)
// Data movement: weight tiles via Tensor Data Mover (tensor_load_to_lds),
// bf16 activation tiles via global_load_async_to_lds_b128, fp32 tiles via
// VALU convert staging. All guarded by __has_builtin with fallbacks.
// ---------------------------------------------------------------------------

typedef __bf16 bf16_t;
typedef __attribute__((ext_vector_type(16))) __bf16 v16bf;
typedef __attribute__((ext_vector_type(8)))  __bf16 v8bf;
typedef __attribute__((ext_vector_type(8)))  float  v8f;
typedef __attribute__((ext_vector_type(4))) unsigned int u32x4;
typedef __attribute__((ext_vector_type(8))) int i32x8;
typedef __attribute__((ext_vector_type(4))) int i32x4;
typedef int v4i_vs __attribute__((vector_size(16)));   // matches builtin param type

#define AS1 __attribute__((address_space(1)))
#define AS3 __attribute__((address_space(3)))

#if __has_builtin(__builtin_amdgcn_tensor_load_to_lds)
#define HAVE_TDM 1
#else
#define HAVE_TDM 0
#endif

#if __has_builtin(__builtin_amdgcn_global_load_async_to_lds_b128)
#define HAVE_ASYNC 1
#else
#define HAVE_ASYNC 0
#endif

#define BM 128
#define BN 128
#define BK 32

// ---- wait helpers ---------------------------------------------------------
__device__ __forceinline__ void wait_async0() {
#if __has_builtin(__builtin_amdgcn_s_wait_asynccnt)
  __builtin_amdgcn_s_wait_asynccnt(0);
#else
  asm volatile("s_wait_asynccnt 0x0" ::: "memory");
#endif
}

// ---- TDM: 2D bf16 tile (tile_d0 x tile_d1 elements) global -> LDS ---------
// Descriptor per CDNA5 ISA 8.3/8.4. Issued once per wave; tracked on TENSORcnt.
__device__ __forceinline__ void tdm_load_2d_bf16(
    const bf16_t* gsrc, unsigned lds_off,
    unsigned tensor_d0, unsigned tensor_d1, unsigned long long stride0,
    unsigned tile_d0, unsigned tile_d1)
{
#if HAVE_TDM
  unsigned long long ga = (unsigned long long)(uintptr_t)gsrc;
  u32x4 g0;
  g0[0] = 1u;                                                 // count=1, user D#
  g0[1] = lds_off;                                            // lds_addr (bytes)
  g0[2] = (unsigned)(ga & 0xffffffffu);                       // global_addr[31:0]
  g0[3] = (unsigned)((ga >> 32) & 0x01ffffffu) | (2u << 30);  // addr[56:32] | type=2
  i32x8 g1;
  g1[0] = (int)(1u << 16);                                    // data_size=1 (2B), wg_mask=0
  g1[1] = (int)((tensor_d0 & 0xffffu) << 16);                 // tensor_dim0[15:0] @ 63:48
  g1[2] = (int)(((tensor_d0 >> 16) & 0xffffu) | ((tensor_d1 & 0xffffu) << 16));
  g1[3] = (int)(((tensor_d1 >> 16) & 0xffffu) | (tile_d0 << 16));   // tile_dim0 @127:112
  g1[4] = (int)(tile_d1 & 0xffffu);                           // tile_dim1, tile_dim2=0
  g1[5] = (int)(unsigned)(stride0 & 0xffffffffu);             // dim0_stride[31:0]
  g1[6] = (int)(unsigned)((stride0 >> 32) & 0xffffu);         // dim0_stride[47:32]
  g1[7] = 0;
  i32x4 gz = {0, 0, 0, 0};
#if __clang_major__ >= 23
  i32x8 gz8 = {0, 0, 0, 0, 0, 0, 0, 0};
  __builtin_amdgcn_tensor_load_to_lds(g0, g1, gz, gz, gz8, 0);
#else
  __builtin_amdgcn_tensor_load_to_lds(g0, g1, gz, gz, 0);
#endif
#else
  (void)gsrc; (void)lds_off; (void)tensor_d0; (void)tensor_d1;
  (void)stride0; (void)tile_d0; (void)tile_d1;
#endif
}

// ---- fp32 -> 16x bf16 convert+store into LDS (two 16B stores) -------------
__device__ __forceinline__ void cvt16_store(bf16_t* dst, const float* src) {
  float4 f0 = ((const float4*)src)[0];
  float4 f1 = ((const float4*)src)[1];
  float4 f2 = ((const float4*)src)[2];
  float4 f3 = ((const float4*)src)[3];
  v8bf lo, hi;
  lo[0] = (__bf16)f0.x; lo[1] = (__bf16)f0.y; lo[2] = (__bf16)f0.z; lo[3] = (__bf16)f0.w;
  lo[4] = (__bf16)f1.x; lo[5] = (__bf16)f1.y; lo[6] = (__bf16)f1.z; lo[7] = (__bf16)f1.w;
  hi[0] = (__bf16)f2.x; hi[1] = (__bf16)f2.y; hi[2] = (__bf16)f2.z; hi[3] = (__bf16)f2.w;
  hi[4] = (__bf16)f3.x; hi[5] = (__bf16)f3.y; hi[6] = (__bf16)f3.z; hi[7] = (__bf16)f3.w;
  ((v8bf*)dst)[0] = lo;
  ((v8bf*)dst)[1] = hi;
}

// ---- build a v16bf fragment from two 16B LDS chunks -----------------------
__device__ __forceinline__ v16bf frag2(const bf16_t* p0, const bf16_t* p1) {
  union U { struct { v8bf a, b; } s; v16bf v; };
  U u;
  u.s.a = *(const v8bf*)p0;
  u.s.b = *(const v8bf*)p1;
  return u.v;
}

// ---------------------------------------------------------------------------
// Generic GEMM: C[M,N] = concatK(A1 bf16 [M,K1], A2 f32 [M,K2]) @ W[N,K1+K2]^T
// Epilogues: 0 = gelu(acc+b) -> bf16      (encoder/decoder hidden)
//            1 = 0.5*acc + 0.5*b -> bf16  (encoder output, boundary folded)
//            2 = acc + b -> f32           (decoder bulk)
//            3 = res + 0.1*acc + 0.1*b -> f32 (error-correct, final output)
// ---------------------------------------------------------------------------
template<int EPI>
__global__ void __launch_bounds__(256) gemm_kernel(
    const bf16_t* __restrict__ A1, int K1,
    const float*  __restrict__ A2, int K2,
    const bf16_t* __restrict__ W,
    const float*  __restrict__ bias,
    const float*  __restrict__ residual,
    float*  __restrict__ outF,
    bf16_t* __restrict__ outB,
    int N)
{
  __shared__ bf16_t sA[BM * BK];
  __shared__ bf16_t sB[BN * BK];

  const int m0 = blockIdx.y * BM;
  const int n0 = blockIdx.x * BN;
  const int Kt = K1 + K2;

  const int t    = threadIdx.x;
  const int lane = t & 31;
  const int wave = t >> 5;
  const int wr   = wave & 3;   // 4 wave-rows  * 32 rows
  const int wc   = wave >> 2;  // 2 wave-cols  * 64 cols
  const int half = lane >> 4;
  const int l15  = lane & 15;

  // staging assignment: 256 threads, 16 elements each (row sr, k-chunk sk)
  const int sr = t >> 1;
  const int sk = (t & 1) << 4;

#if HAVE_TDM
  const unsigned sB_off = (unsigned)(uintptr_t)(void*)sB;
#endif

  v8f acc[2][4];
#pragma unroll
  for (int r = 0; r < 2; ++r)
#pragma unroll
    for (int c = 0; c < 4; ++c)
#pragma unroll
      for (int i = 0; i < 8; ++i) acc[r][c][i] = 0.0f;

  for (int k0 = 0; k0 < Kt; k0 += BK) {
    // ---- B tile: Tensor Data Mover DMA (wave 0) or per-thread loads ------
#if HAVE_TDM
    if (wave == 0) {
      tdm_load_2d_bf16(W + (size_t)n0 * Kt + k0, sB_off,
                       (unsigned)Kt, (unsigned)N, (unsigned long long)Kt,
                       (unsigned)BK, (unsigned)BN);
    }
#else
    {
      const bf16_t* src = W + (size_t)(n0 + sr) * Kt + k0 + sk;
      bf16_t* dst = sB + sr * BK + sk;
      ((v8bf*)dst)[0] = ((const v8bf*)src)[0];
      ((v8bf*)dst)[1] = ((const v8bf*)src)[1];
      if (k0 + BK < Kt) __builtin_prefetch(src + BK, 0, 3);
    }
#endif

    // ---- A tile: uniform source select (K1 is a multiple of BK) ----------
    if (k0 < K1) {
      const bf16_t* src = A1 + (size_t)(m0 + sr) * K1 + k0 + sk;
      bf16_t* dst = sA + sr * BK + sk;
#if HAVE_ASYNC
      __builtin_amdgcn_global_load_async_to_lds_b128(
          (AS1 v4i_vs*)(void*)src, (AS3 v4i_vs*)(void*)dst, 0, 0);
      __builtin_amdgcn_global_load_async_to_lds_b128(
          (AS1 v4i_vs*)(void*)src, (AS3 v4i_vs*)(void*)dst, 16, 0);
#else
      ((v8bf*)dst)[0] = ((const v8bf*)src)[0];
      ((v8bf*)dst)[1] = ((const v8bf*)src)[1];
      if (k0 + BK < K1) __builtin_prefetch(src + BK, 0, 3);
#endif
    } else {
      const float* src = A2 + (size_t)(m0 + sr) * K2 + (k0 + sk - K1);
      cvt16_store(sA + sr * BK + sk, src);
      if (k0 + BK < Kt) __builtin_prefetch(src + BK, 0, 3);
    }

    // ---- drain async/tensor counters, then rendezvous --------------------
#if HAVE_ASYNC
    wait_async0();
#endif
#if HAVE_TDM
    if (wave == 0) __builtin_amdgcn_s_wait_tensorcnt(0);
#endif
    __syncthreads();

    // ---- fragments per ISA 7.12.2 layouts --------------------------------
    v16bf afrag[2], bfrag[4];
#pragma unroll
    for (int r = 0; r < 2; ++r) {
      // A 16x32: lane row = l15, K halves at {8*half, 16+8*half}
      const bf16_t* p = sA + (wr * 32 + r * 16 + l15) * BK + 8 * half;
      afrag[r] = frag2(p, p + 16);
    }
#pragma unroll
    for (int c = 0; c < 4; ++c) {
      // B 32x16: lane col = l15, K halves at {8*half, 16+8*half}
      const bf16_t* p = sB + (wc * 64 + c * 16 + l15) * BK + 8 * half;
      bfrag[c] = frag2(p, p + 16);
    }

#pragma unroll
    for (int r = 0; r < 2; ++r)
#pragma unroll
      for (int c = 0; c < 4; ++c)
        acc[r][c] = __builtin_amdgcn_wmma_f32_16x16x32_bf16(
            false, afrag[r], false, bfrag[c], (short)0, acc[r][c], false, false);

    __syncthreads();
  }

  // ---- epilogue: C layout = VGPR i -> row 8*half + i, col = l15 ----------
#pragma unroll
  for (int c = 0; c < 4; ++c) {
    const int n  = n0 + wc * 64 + c * 16 + l15;
    const float bn = bias[n];
#pragma unroll
    for (int r = 0; r < 2; ++r) {
      const int mb = m0 + wr * 32 + r * 16 + 8 * half;
#pragma unroll
      for (int i = 0; i < 8; ++i) {
        const int m = mb + i;
        float v = acc[r][c][i];
        if constexpr (EPI == 0) {
          v += bn;
          const float g = 0.5f * v * (1.0f + erff(v * 0.70710678118654752f));
          outB[(size_t)m * N + n] = (bf16_t)g;
        } else if constexpr (EPI == 1) {
          outB[(size_t)m * N + n] = (bf16_t)(0.5f * v + 0.5f * bn);
        } else if constexpr (EPI == 2) {
          outF[(size_t)m * N + n] = v + bn;
        } else {
          outF[(size_t)m * N + n] =
              residual[(size_t)m * N + n] + 0.1f * v + 0.1f * bn;
        }
      }
    }
  }
}

// ---------------------------------------------------------------------------
// Weight conversion: f32 [N,K] -> bf16 placed at dst[n*dld + koff + k]
// ---------------------------------------------------------------------------
__global__ void __launch_bounds__(256) conv_weight(
    bf16_t* __restrict__ dst, int dld, int koff,
    const float* __restrict__ src, int N, int K)
{
  const int idx = blockIdx.x * 256 + threadIdx.x;
  if (idx >= N * K) return;
  const int n = idx / K, k = idx - n * K;
  dst[(size_t)n * dld + koff + k] = (bf16_t)src[idx];
}

// proj pre-scaled by S * cos(phase[n]);  S = sum_i prop_i * softmax(shell)_i
__global__ void __launch_bounds__(256) conv_proj(
    bf16_t* __restrict__ dst, int dld, int koff,
    const float* __restrict__ src,
    const float* __restrict__ phase,
    const float* __restrict__ shell,
    int N, int K)
{
  const int idx = blockIdx.x * 256 + threadIdx.x;
  if (idx >= N * K) return;
  const float tt[4] = {-2.0f, -0.66666666667f, 0.66666666667f, 2.0f};
  float es[4], se = 0.0f;
#pragma unroll
  for (int i = 0; i < 4; ++i) { es[i] = expf(shell[i]); se += es[i]; }
  float S = 0.0f;
#pragma unroll
  for (int i = 0; i < 4; ++i) {
    const float z2 = expf(2.0f * tt[i]);          // z^DELTA, DELTA=2
    const float prop = z2 / ((1.0f + z2) * (1.0f + z2) + 1e-8f);
    S += prop * es[i] / se;
  }
  const int n = idx / K, k = idx - n * K;
  dst[(size_t)n * dld + koff + k] = (bf16_t)(src[idx] * S * cosf(phase[n]));
}

// ---------------------------------------------------------------------------
extern "C" void kernel_launch(void* const* d_in, const int* in_sizes, int n_in,
                              void* d_out, int out_size, void* d_ws, size_t ws_size,
                              hipStream_t stream)
{
  (void)in_sizes; (void)n_in; (void)out_size; (void)ws_size;
  const int M = 4 * 4096;          // B*L tokens
  const int D = 2048, HENC = 1024, BD = 512, HDEC = 1024;

  char* ws = (char*)d_ws;
  size_t off = 0;
  auto alloc = [&](size_t bytes) -> void* {
    void* p = ws + off;
    off = (off + bytes + 255) & ~(size_t)255;
    return p;
  };

  bf16_t* Wenc1 = (bf16_t*)alloc((size_t)HENC * D * 2);
  bf16_t* Wenc2 = (bf16_t*)alloc((size_t)BD * (HENC + D) * 2);  // [w2 | proj'] K=3072
  bf16_t* Wdec1 = (bf16_t*)alloc((size_t)HDEC * BD * 2);
  bf16_t* Wdec2 = (bf16_t*)alloc((size_t)D * HDEC * 2);
  bf16_t* Wec   = (bf16_t*)alloc((size_t)D * D * 2);
  bf16_t* Hbuf  = (bf16_t*)alloc((size_t)M * HENC * 2);         // reused enc/dec hidden
  bf16_t* Bnd   = (bf16_t*)alloc((size_t)M * BD * 2);
  float*  Bulk  = (float*) alloc((size_t)M * D * 4);

  float* out = (float*)d_out;
  const dim3 blk(256);
  auto cgrid = [](int total) { return dim3((unsigned)((total + 255) / 256)); };

  for (int b = 0; b < 2; ++b) {     // 0 = key branch, 1 = value branch
    const float* X   = (const float*)d_in[b];
    const float* ew1 = (const float*)d_in[2 + 7 * b + 0];
    const float* eb1 = (const float*)d_in[2 + 7 * b + 1];
    const float* ew2 = (const float*)d_in[2 + 7 * b + 2];
    const float* eb2 = (const float*)d_in[2 + 7 * b + 3];
    const float* epj = (const float*)d_in[2 + 7 * b + 4];
    const float* esh = (const float*)d_in[2 + 7 * b + 5];
    const float* eph = (const float*)d_in[2 + 7 * b + 6];
    const float* dw1 = (const float*)d_in[16 + 6 * b + 0];
    const float* db1 = (const float*)d_in[16 + 6 * b + 1];
    const float* dw2 = (const float*)d_in[16 + 6 * b + 2];
    const float* db2 = (const float*)d_in[16 + 6 * b + 3];
    const float* ecw = (const float*)d_in[16 + 6 * b + 4];
    const float* ecb = (const float*)d_in[16 + 6 * b + 5];

    // weight conversion (stream-ordered; buffers reused across branches)
    conv_weight<<<cgrid(HENC * D),  blk, 0, stream>>>(Wenc1, D, 0, ew1, HENC, D);
    conv_weight<<<cgrid(BD * HENC), blk, 0, stream>>>(Wenc2, HENC + D, 0, ew2, BD, HENC);
    conv_proj  <<<cgrid(BD * D),    blk, 0, stream>>>(Wenc2, HENC + D, HENC, epj, eph, esh, BD, D);
    conv_weight<<<cgrid(HDEC * BD), blk, 0, stream>>>(Wdec1, BD, 0, dw1, HDEC, BD);
    conv_weight<<<cgrid(D * HDEC),  blk, 0, stream>>>(Wdec2, HDEC, 0, dw2, D, HDEC);
    conv_weight<<<cgrid(D * D),     blk, 0, stream>>>(Wec,   D, 0, ecw, D, D);

    // E1: H = gelu(X @ w1^T + b1)
    gemm_kernel<0><<<dim3(HENC / BN, M / BM), blk, 0, stream>>>(
        nullptr, 0, X, D, Wenc1, eb1, nullptr, nullptr, Hbuf, HENC);
    // E2: bnd = 0.5*(H@w2^T + b2) + 0.5*S*cos(ph)*(X@proj^T)  (K concat = 3072)
    gemm_kernel<1><<<dim3(BD / BN, M / BM), blk, 0, stream>>>(
        Hbuf, HENC, X, D, Wenc2, eb2, nullptr, nullptr, Bnd, BD);
    // D1: H = gelu(bnd @ dw1^T + db1)
    gemm_kernel<0><<<dim3(HDEC / BN, M / BM), blk, 0, stream>>>(
        Bnd, BD, nullptr, 0, Wdec1, db1, nullptr, nullptr, Hbuf, HDEC);
    // D2: bulk = H @ dw2^T + db2
    gemm_kernel<2><<<dim3(D / BN, M / BM), blk, 0, stream>>>(
        Hbuf, HDEC, nullptr, 0, Wdec2, db2, nullptr, Bulk, nullptr, D);
    // D3: out = bulk + 0.1*(bulk @ ec^T + ec_b)
    gemm_kernel<3><<<dim3(D / BN, M / BM), blk, 0, stream>>>(
        nullptr, 0, Bulk, D, Wec, ecb, Bulk, out + (size_t)b * M * D, nullptr, D);
  }
}
